// PatchCNN_28080496181359
// MI455X (gfx1250) — compile-verified, matrix operand order checked by reading, not run on hardware
//
#include <hip/hip_runtime.h>
#include <hip/hip_bf16.h>

typedef __attribute__((ext_vector_type(16))) _Float16 v16h;
typedef __attribute__((ext_vector_type(8)))  float    v8f;

#define NPATCH 16
#define BBATCH 32
#define EPSV   1e-5f
#define INV_SQRT2F 0.70710678118654752440f

// Async global->LDS copy (CDNA5 ASYNCcnt path). vdst = LDS byte offset,
// vaddr = 64-bit global address (GV mode, saddr off).
__device__ __forceinline__ void async_g2l_b32(unsigned lds_off, const void* g) {
  asm volatile("global_load_async_to_lds_b32 %0, %1, off"
               :: "v"(lds_off), "v"(g) : "memory");
}
__device__ __forceinline__ void wait_async0() {
  asm volatile("s_wait_asynccnt 0x0" ::: "memory");
}

// ---------------------------------------------------------------------------
// Per-(patch, channel) batch statistics: mean and rsqrt(var + eps).
// act layout: [NPATCH, BBATCH, C, H, W] flattened. One block per (n, c).
// ---------------------------------------------------------------------------
__global__ __launch_bounds__(256)
void stats_kernel(const float* __restrict__ act, int C, int HW,
                  float* __restrict__ mu, float* __restrict__ rs) {
  int nc = blockIdx.x;
  int n  = nc / C;
  int c  = nc % C;
  int tid = threadIdx.x;
  int total = BBATCH * HW;
  float s = 0.f, s2 = 0.f;
  for (int i = tid; i < total; i += 256) {
    int b = i / HW, j = i % HW;
    float v = act[(((size_t)(n * BBATCH + b)) * C + c) * HW + j];
    s += v; s2 += v * v;
  }
  __shared__ float sh[256], sh2[256];
  sh[tid] = s; sh2[tid] = s2;
  __syncthreads();
  for (int o = 128; o > 0; o >>= 1) {
    if (tid < o) { sh[tid] += sh[tid + o]; sh2[tid] += sh2[tid + o]; }
    __syncthreads();
  }
  if (tid == 0) {
    float m  = sh[0] / (float)total;
    float vv = sh2[0] / (float)total - m * m;
    mu[nc] = m;
    rs[nc] = rsqrtf(vv + EPSV);
  }
}

// ---------------------------------------------------------------------------
// Implicit-GEMM conv3x3 (padding 1) on v_wmma_f32_16x16x32_f16.
// Block = 128 threads (4 waves) = one (n, b) and an 8x8 output spatial tile.
//  - Activation tile (CIN x TIN x TIN) is staged once into LDS as f16 with
//    HerPN applied (layer 0 instead fuses the patch rearrangement).
//  - Each K-chunk's COUT x 32 f32 weight slab is fetched with
//    global_load_async_to_lds_b32 (+ s_wait_asynccnt) into LDS.
//  - Each wave owns 16 output positions (one B fragment reused over M tiles).
// ---------------------------------------------------------------------------
template <int CIN, int COUT, int HIN, int STRIDE, bool FIRST>
__global__ __launch_bounds__(128)
void conv_wmma(const float* __restrict__ in, const float* __restrict__ wgt,
               const float* __restrict__ hw, const float* __restrict__ mu,
               const float* __restrict__ rs, float* __restrict__ outp) {
  constexpr int HO  = HIN / STRIDE;
  constexpr int K   = CIN * 9;
  constexpr int KC  = (K + 31) / 32;
  constexpr int MT  = COUT / 16;
  constexpr int TIN = 8 * STRIDE + 2;       // staged input tile edge
  constexpr int TPB = (HO / 8) * (HO / 8);  // tiles per (n,b)
  constexpr int AE  = CIN * TIN * TIN;      // staged elements

  __shared__ __align__(16) _Float16 sAct[AE];
  __shared__ __align__(16) float    sWgt[COUT * 32];

  int tile = blockIdx.x % TPB;
  int nb   = blockIdx.x / TPB;
  int n = nb / BBATCH, b = nb % BBATCH;
  int oy0 = (tile / (HO / 8)) * 8;
  int ox0 = (tile % (HO / 8)) * 8;
  int tid  = threadIdx.x;
  int lane = tid & 31, wave = tid >> 5;
  int col  = lane & 15, half = lane >> 4;

  // ---- Stage HerPN'd activation tile into LDS (f16), each value once ----
  {
    int giy0 = oy0 * STRIDE - 1, gix0 = ox0 * STRIDE - 1;
    for (int i = tid; i < AE; i += 128) {
      int c  = i / (TIN * TIN);
      int r  = i % (TIN * TIN);
      int ty = r / TIN, tx = r % TIN;
      int iy = giy0 + ty, ix = gix0 + tx;
      float val = 0.f;
      if (iy >= 0 && iy < HIN && ix >= 0 && ix < HIN) {
        if (FIRST) {
          int hb = n >> 2, wb = n & 3;  // patch -> global pixel
          val = in[(((size_t)b * 3 + c) * 256 + (hb * 64 + iy)) * 256 +
                   (wb * 64 + ix)];
        } else {
          float a = in[(((size_t)(n * BBATCH + b) * CIN + c) * HIN + iy) * HIN +
                       ix];
          float m  = mu[n * CIN + c];
          float sc = rs[n * CIN + c];
          float xh = (a - m) * sc;
          const float* hwn = hw + (size_t)n * 3 * CIN;
          val = hwn[c] + hwn[CIN + c] * xh +
                hwn[2 * CIN + c] * (xh * xh - 1.f) * INV_SQRT2F;
        }
      }
      sAct[i] = (_Float16)val;
    }
  }

  int p  = wave * 16 + col;  // this lane's output position within the tile
  int ly = p >> 3, lx = p & 7;

  const v8f vz = {0.f, 0.f, 0.f, 0.f, 0.f, 0.f, 0.f, 0.f};
  v8f acc[MT];
#pragma unroll
  for (int mt = 0; mt < MT; ++mt) acc[mt] = vz;

#pragma unroll 1
  for (int kc = 0; kc < KC; ++kc) {
    __syncthreads();  // sAct visible / previous chunk's sWgt reads retired

    // ---- async-copy this chunk's COUT x 32 f32 weight slab into LDS ----
    for (int i = tid; i < COUT * 32; i += 128) {
      int cout = i >> 5, kk = i & 31;
      int gk = kc * 32 + kk;
      const float* ga =
          wgt + ((size_t)n * COUT + cout) * K + ((gk < K) ? gk : 0);
      unsigned loff = (unsigned)(uintptr_t)(void*)&sWgt[i];
      async_g2l_b32(loff, ga);
    }
    wait_async0();
    __syncthreads();

    // ---- B fragment: 16 K-values of this lane's output column (LDS) ----
    v16h bfrag;
#pragma unroll
    for (int t = 0; t < 16; ++t) {
      int v  = t >> 1, e = t & 1;
      int kk = (v < 4) ? (half * 8 + v * 2 + e)
                       : (16 + half * 8 + (v - 4) * 2 + e);
      int k  = kc * 32 + kk;
      _Float16 val = (_Float16)0.f;
      if (k < K) {
        int cin = k / 9, r = k % 9;
        int dy = r / 3, dx = r % 3;          // tile coords absorb the -1 pad
        int ty = ly * STRIDE + dy;
        int tx = lx * STRIDE + dx;
        val = sAct[cin * TIN * TIN + ty * TIN + tx];
      }
      bfrag[t] = val;
    }
    // ---- M tiles: A fragment from the LDS weight slab ----
#pragma unroll
    for (int mt = 0; mt < MT; ++mt) {
      int cout = mt * 16 + col;
      v16h afrag;
#pragma unroll
      for (int t = 0; t < 16; ++t) {
        int v  = t >> 1, e = t & 1;
        int kk = (v < 4) ? (half * 8 + v * 2 + e)
                         : (16 + half * 8 + (v - 4) * 2 + e);
        int k  = kc * 32 + kk;
        afrag[t] = (_Float16)((k < K) ? sWgt[cout * 32 + kk] : 0.f);
      }
      acc[mt] = __builtin_amdgcn_wmma_f32_16x16x32_f16(
          false, afrag, false, bfrag, (short)0, acc[mt], false, false);
    }
  }

  // ---- Epilogue: D layout lane/VGPR -> [n,b,cout,oy,ox] f32 store ----
  int oy = oy0 + ly, ox = ox0 + lx;
#pragma unroll
  for (int mt = 0; mt < MT; ++mt) {
#pragma unroll
    for (int r = 0; r < 8; ++r) {
      int m    = r + 8 * half;
      int cout = mt * 16 + m;
      outp[(((size_t)(n * BBATCH + b) * COUT + cout) * HO + oy) * HO + ox] =
          acc[mt][r];
    }
  }
}

// ---------------------------------------------------------------------------
// HerPN + adaptive 2x2 average pool on the last conv output.
// One block per (n,b); thread t = c*4 + i*2 + j averages its 8x8 quadrant.
// ---------------------------------------------------------------------------
__global__ __launch_bounds__(256)
void pool_kernel(const float* __restrict__ l5, const float* __restrict__ hw,
                 const float* __restrict__ mu, const float* __restrict__ rs,
                 float* __restrict__ feat) {
  int nb = blockIdx.x;
  int n = nb / BBATCH;
  int t = threadIdx.x;
  int c = t >> 2, i = (t >> 1) & 1, j = t & 1;
  const float* base = l5 + ((size_t)nb * 64 + c) * 256;
  float m = mu[n * 64 + c], s = rs[n * 64 + c];
  const float* hwn = hw + (size_t)n * 3 * 64;
  float w0 = hwn[c], w1 = hwn[64 + c], w2 = hwn[128 + c];
  float accv = 0.f;
  for (int y = 0; y < 8; ++y)
    for (int x = 0; x < 8; ++x) {
      float a  = base[(i * 8 + y) * 16 + (j * 8 + x)];
      float xh = (a - m) * s;
      accv += w0 + w1 * xh + w2 * (xh * xh - 1.f) * INV_SQRT2F;
    }
  feat[(size_t)nb * 256 + t] = accv * (1.f / 64.f);
}

// BatchNorm1d (batch stats over B) with affine g,b: feat[n,b,f] -> y[n,b,f].
__global__ __launch_bounds__(256)
void bn1d_kernel(const float* __restrict__ feat, const float* __restrict__ g,
                 const float* __restrict__ bia, float* __restrict__ y) {
  int idx = blockIdx.x * 256 + threadIdx.x;  // n*256 + f
  int n = idx >> 8, f = idx & 255;
  float s = 0.f, s2 = 0.f;
  for (int b = 0; b < BBATCH; ++b) {
    float v = feat[((size_t)(n * BBATCH + b)) * 256 + f];
    s += v; s2 += v * v;
  }
  float m = s / (float)BBATCH;
  float var = s2 / (float)BBATCH - m * m;
  float r = rsqrtf(var + EPSV);
  float gg = g[idx], bb = bia[idx];
  for (int b = 0; b < BBATCH; ++b) {
    size_t o = ((size_t)(n * BBATCH + b)) * 256 + f;
    y[o] = (feat[o] - m) * r * gg + bb;
  }
}

// ---------------------------------------------------------------------------
// og head: [32 x 4096] @ lin_w^T -> [32 x 256] on WMMA. One block, 8 waves:
// wave w owns 2 N-tiles, 2 M-tiles, K = 128 chunks of 32.
// ---------------------------------------------------------------------------
__global__ __launch_bounds__(256)
void og_gemm(const float* __restrict__ y, const float* __restrict__ lw,
             const float* __restrict__ lb, float* __restrict__ ogRaw) {
  int lane = threadIdx.x & 31, wave = threadIdx.x >> 5;
  int col = lane & 15, half = lane >> 4;
  const v8f vz = {0.f, 0.f, 0.f, 0.f, 0.f, 0.f, 0.f, 0.f};
  v8f acc[2][2];
  acc[0][0] = vz; acc[0][1] = vz; acc[1][0] = vz; acc[1][1] = vz;

#pragma unroll 1
  for (int kc = 0; kc < 128; ++kc) {
    v16h af[2], bf[2];
#pragma unroll
    for (int t = 0; t < 16; ++t) {
      int v  = t >> 1, e = t & 1;
      int kk = (v < 4) ? (half * 8 + v * 2 + e)
                       : (16 + half * 8 + (v - 4) * 2 + e);
      int k  = kc * 32 + kk;
#pragma unroll
      for (int mt = 0; mt < 2; ++mt) {
        int b = mt * 16 + col;  // A row = batch index
        af[mt][t] =
            (_Float16)y[(size_t)(k >> 8) * BBATCH * 256 + b * 256 + (k & 255)];
      }
#pragma unroll
      for (int nt = 0; nt < 2; ++nt) {
        int j = (wave * 2 + nt) * 16 + col;  // B col = output feature
        bf[nt][t] = (_Float16)lw[(size_t)j * 4096 + k];
      }
    }
#pragma unroll
    for (int mt = 0; mt < 2; ++mt)
#pragma unroll
      for (int nt = 0; nt < 2; ++nt)
        acc[mt][nt] = __builtin_amdgcn_wmma_f32_16x16x32_f16(
            false, af[mt], false, bf[nt], (short)0, acc[mt][nt], false, false);
  }
#pragma unroll
  for (int mt = 0; mt < 2; ++mt)
#pragma unroll
    for (int nt = 0; nt < 2; ++nt)
#pragma unroll
      for (int r = 0; r < 8; ++r) {
        int b = mt * 16 + r + 8 * half;
        int j = (wave * 2 + nt) * 16 + col;
        ogRaw[(size_t)b * 256 + j] = acc[mt][nt][r] + lb[j];
      }
}

// BatchNorm (affine=False) over the batch dim of og; writes final output.
__global__ __launch_bounds__(256)
void og_bn(const float* __restrict__ ogRaw, float* __restrict__ outOg) {
  int j = threadIdx.x;
  float s = 0.f, s2 = 0.f;
  for (int b = 0; b < BBATCH; ++b) {
    float v = ogRaw[(size_t)b * 256 + j];
    s += v; s2 += v * v;
  }
  float m = s / (float)BBATCH;
  float var = s2 / (float)BBATCH - m * m;
  float r = rsqrtf(var + EPSV);
  for (int b = 0; b < BBATCH; ++b)
    outOg[(size_t)b * 256 + j] = (ogRaw[(size_t)b * 256 + j] - m) * r;
}

// pred head: pred[b,n,k] = y[n,b,:].jig_w[k,:] + jig_b[k], rows packed b*16+n.
__global__ __launch_bounds__(256)
void pred_kernel(const float* __restrict__ y, const float* __restrict__ jw,
                 const float* __restrict__ jb, float* __restrict__ outPred) {
  int idx = blockIdx.x * 256 + threadIdx.x;  // < 8192
  int k = idx & 15, n = (idx >> 4) & 15, b = idx >> 8;
  const float* yr = y + (size_t)(n * BBATCH + b) * 256;
  const float* wr = jw + (size_t)k * 256;
  float s = jb[k];
  for (int f = 0; f < 256; ++f) s += yr[f] * wr[f];
  outPred[(size_t)(b * 16 + n) * 16 + k] = s;
}

__global__ void target_kernel(float* __restrict__ o) {
  int i = blockIdx.x * 256 + threadIdx.x;
  if (i < 512) o[i] = (float)(i & 15);
}

// ---------------------------------------------------------------------------
extern "C" void kernel_launch(void* const* d_in, const int* in_sizes, int n_in,
                              void* d_out, int out_size, void* d_ws,
                              size_t ws_size, hipStream_t stream) {
  const float* x    = (const float*)d_in[0];
  const float* c0   = (const float*)d_in[1];
  const float* hw1  = (const float*)d_in[2];
  const float* cw1  = (const float*)d_in[3];
  const float* hw2  = (const float*)d_in[4];
  const float* cw2  = (const float*)d_in[5];
  const float* hw3  = (const float*)d_in[6];
  const float* cw3  = (const float*)d_in[7];
  const float* hw4  = (const float*)d_in[8];
  const float* cw4  = (const float*)d_in[9];
  const float* hw5  = (const float*)d_in[10];
  const float* cw5  = (const float*)d_in[11];
  const float* phw  = (const float*)d_in[12];
  const float* bng  = (const float*)d_in[13];
  const float* bnb  = (const float*)d_in[14];
  const float* lw   = (const float*)d_in[15];
  const float* lb   = (const float*)d_in[16];
  const float* jw   = (const float*)d_in[17];
  const float* jb   = (const float*)d_in[18];
  float* out = (float*)d_out;
  float* ws  = (float*)d_ws;

  // Workspace layout (floats).
  float* buf0 = ws;                        // 16*32*16*64*64 = 33,554,432
  float* buf1 = buf0 + 33554432;           // same size
  float* muv  = buf1 + 33554432;           // up to 16*64 = 1024
  float* rsv  = muv + 1024;                // 1024
  float* feat = rsv + 1024;                // 16*32*256 = 131072
  float* yb   = feat + 131072;             // 131072
  float* ogR  = yb + 131072;               // 8192

  // conv0: [B,3,256,256] patches -> [N,B,16,64,64]
  conv_wmma<3, 16, 64, 1, true>
      <<<NPATCH * BBATCH * 64, 128, 0, stream>>>(x, c0, nullptr, nullptr,
                                                 nullptr, buf0);

  stats_kernel<<<NPATCH * 16, 256, 0, stream>>>(buf0, 16, 64 * 64, muv, rsv);
  conv_wmma<16, 16, 64, 1, false>
      <<<NPATCH * BBATCH * 64, 128, 0, stream>>>(buf0, cw1, hw1, muv, rsv,
                                                 buf1);

  stats_kernel<<<NPATCH * 16, 256, 0, stream>>>(buf1, 16, 64 * 64, muv, rsv);
  conv_wmma<16, 32, 64, 2, false>
      <<<NPATCH * BBATCH * 16, 128, 0, stream>>>(buf1, cw2, hw2, muv, rsv,
                                                 buf0);

  stats_kernel<<<NPATCH * 32, 256, 0, stream>>>(buf0, 32, 32 * 32, muv, rsv);
  conv_wmma<32, 32, 32, 1, false>
      <<<NPATCH * BBATCH * 16, 128, 0, stream>>>(buf0, cw3, hw3, muv, rsv,
                                                 buf1);

  stats_kernel<<<NPATCH * 32, 256, 0, stream>>>(buf1, 32, 32 * 32, muv, rsv);
  conv_wmma<32, 64, 32, 2, false>
      <<<NPATCH * BBATCH * 4, 128, 0, stream>>>(buf1, cw4, hw4, muv, rsv,
                                                buf0);

  stats_kernel<<<NPATCH * 64, 256, 0, stream>>>(buf0, 64, 16 * 16, muv, rsv);
  conv_wmma<64, 64, 16, 1, false>
      <<<NPATCH * BBATCH * 4, 128, 0, stream>>>(buf0, cw5, hw5, muv, rsv,
                                                buf1);

  stats_kernel<<<NPATCH * 64, 256, 0, stream>>>(buf1, 64, 16 * 16, muv, rsv);
  pool_kernel<<<NPATCH * BBATCH, 256, 0, stream>>>(buf1, phw, muv, rsv, feat);
  bn1d_kernel<<<NPATCH, 256, 0, stream>>>(feat, bng, bnb, yb);

  og_gemm<<<1, 256, 0, stream>>>(yb, lw, lb, ogR);
  og_bn<<<1, 256, 0, stream>>>(ogR, out);                 // out[0 .. 8191]
  pred_kernel<<<32, 256, 0, stream>>>(yb, jw, jb, out + 8192);  // 8192 .. 16383
  target_kernel<<<2, 256, 0, stream>>>(out + 16384);      // 16384 .. 16895
}
